// MultiHeadAttention_54872502174451
// MI455X (gfx1250) — matrix operand order repaired and checked by reading, not compiled, not run on hardware
//
#include <hip/hip_runtime.h>
#include <hip/hip_bf16.h>
#include <stdint.h>

#define Bb 2
#define Ss 2048
#define Ee 1024
#define Hh 16
#define Dd 64

typedef __attribute__((ext_vector_type(16))) __bf16 v16bf;
typedef __attribute__((ext_vector_type(8)))  float  v8f;
typedef __attribute__((ext_vector_type(4)))  float  v4f;
typedef __attribute__((ext_vector_type(16))) unsigned short v16u;
typedef __attribute__((ext_vector_type(8)))  unsigned short v8u;
typedef __attribute__((ext_vector_type(4)))  unsigned short v4u;
typedef int gcc_v4i __attribute__((vector_size(16)));

#if defined(__AMDGCN__) && __has_builtin(__builtin_amdgcn_global_load_async_to_lds_b128)
#define HAS_ASYNC_LDS 1
#else
#define HAS_ASYNC_LDS 0
#endif

__device__ __forceinline__ unsigned short bf16_rtne(float f) {
  uint32_t u = __builtin_bit_cast(uint32_t, f);
  return (unsigned short)((u + 0x7FFFu + ((u >> 16) & 1u)) >> 16);
}

__device__ __forceinline__ v16u cat16(v8u lo, v8u hi) {
  return __builtin_shufflevector(lo, hi, 0, 1, 2, 3, 4, 5, 6, 7,
                                          8, 9, 10, 11, 12, 13, 14, 15);
}

__device__ __forceinline__ v8f wmma_bf16(v16u a, v16u b, v8f c) {
  return __builtin_amdgcn_wmma_f32_16x16x32_bf16(
      false, __builtin_bit_cast(v16bf, a),
      false, __builtin_bit_cast(v16bf, b),
      (short)0, c, false, false);
}

// 16-byte global -> LDS copy. Async (ASYNCcnt) when the toolchain exposes the
// gfx1250 builtin; otherwise synchronous VGPR round-trip.
__device__ __forceinline__ void cp16_g2l(const unsigned short* g, unsigned short* l) {
#if HAS_ASYNC_LDS
  __builtin_amdgcn_global_load_async_to_lds_b128(
      (__attribute__((address_space(1))) gcc_v4i*)(uintptr_t)g,
      (__attribute__((address_space(3))) gcc_v4i*)(uint32_t)(uintptr_t)l, 0, 0);
#else
  *(v8u*)l = *(const v8u*)g;
#endif
}

__device__ __forceinline__ void wait_cp() {
#if defined(__AMDGCN__) && HAS_ASYNC_LDS
#if __has_builtin(__builtin_amdgcn_s_wait_asynccnt)
  __builtin_amdgcn_s_wait_asynccnt(0);
#else
  asm volatile("s_wait_asynccnt 0x0" ::: "memory");
#endif
#endif
}

// Streaming f32 -> bf16 (rtne) conversion, 4 elements/thread.
__global__ __launch_bounds__(256)
void cvt_f32_bf16(const float* __restrict__ src, unsigned short* __restrict__ dst,
                  int n4) {
  const int i = blockIdx.x * 256 + threadIdx.x;
  if (i < n4) {
    v4f x = *(const v4f*)(src + (size_t)i * 4);
    v4u o;
    o.x = bf16_rtne(x.x); o.y = bf16_rtne(x.y);
    o.z = bf16_rtne(x.z); o.w = bf16_rtne(x.w);
    *(v4u*)(dst + (size_t)i * 4) = o;
  }
}

// OUT = relu(X @ W^T + bias); X:[M][K] bf16 bits, W:[N][K] bf16 bits.
// OUT_MODE 0: bf16 bits [M][N]
// OUT_MODE 1: bf16 bits, V-transposed per head: [B][H][D][S]
// OUT_MODE 2: f32 [M][N]
// 128x128 block, 8 waves, K-step 64, double-buffered (async) LDS staging.
template <int OUT_MODE>
__global__ __launch_bounds__(256)
void proj_gemm(const unsigned short* __restrict__ X,
               const unsigned short* __restrict__ W,
               const float* __restrict__ bias, void* __restrict__ Outv) {
  constexpr int K = Ee, N = Ee;
  constexpr int KS = 64;
  constexpr int LDT = 72;  // 64 data + 8 pad shorts (144B row) -> conflict-free frags
  __shared__ unsigned short As[2][128 * LDT];
  __shared__ unsigned short Bs[2][128 * LDT];

  const int tid  = threadIdx.x;
  const int lane = tid & 31;
  const int wave = tid >> 5;
  const int wm   = wave >> 2;   // 0..1  (64 rows)
  const int wn   = wave & 3;    // 0..3  (32 cols)
  const int bm   = blockIdx.x * 128;
  const int bn   = blockIdx.y * 128;
  const int lrow = lane & 15;
  const int lkg  = lane >> 4;

  // Stage one 128x64 bf16 panel of X and W into buffer `buf`.
  auto stage = [&](int buf, int k0) {
    #pragma unroll
    for (int i = 0; i < 4; ++i) {
      const int c   = tid + i * 256;   // 1024 16B slots per tensor
      const int row = c >> 3;
      const int kc  = (c & 7) * 8;
      cp16_g2l(X + (size_t)(bm + row) * K + k0 + kc, &As[buf][row * LDT + kc]);
      cp16_g2l(W + (size_t)(bn + row) * K + k0 + kc, &Bs[buf][row * LDT + kc]);
    }
  };

  v8f acc[4][2] = {};
  stage(0, 0);
  int cur = 0;
  for (int k0 = 0; k0 < K; k0 += KS) {
    wait_cp();
    __syncthreads();                       // staged panel visible to all waves
    if (k0 + KS < K) stage(cur ^ 1, k0 + KS);  // prefetch next panel

    #pragma unroll
    for (int ks = 0; ks < 2; ++ks) {
      v16u afr[4], bfr[2];
      #pragma unroll
      for (int mt = 0; mt < 4; ++mt) {
        const unsigned short* p =
            &As[cur][(wm * 64 + mt * 16 + lrow) * LDT + ks * 32 + lkg * 8];
        afr[mt] = cat16(*(const v8u*)p, *(const v8u*)(p + 16));
      }
      #pragma unroll
      for (int nt = 0; nt < 2; ++nt) {
        const unsigned short* p =
            &Bs[cur][(wn * 32 + nt * 16 + lrow) * LDT + ks * 32 + lkg * 8];
        bfr[nt] = cat16(*(const v8u*)p, *(const v8u*)(p + 16));
      }
      #pragma unroll
      for (int mt = 0; mt < 4; ++mt)
        #pragma unroll
        for (int nt = 0; nt < 2; ++nt)
          acc[mt][nt] = wmma_bf16(afr[mt], bfr[nt], acc[mt][nt]);
    }
    cur ^= 1;
  }

  unsigned short* Oh = (unsigned short*)Outv;
  float* Of          = (float*)Outv;
  #pragma unroll
  for (int nt = 0; nt < 2; ++nt) {
    const int n    = bn + wn * 32 + nt * 16 + lrow;
    const float bv = bias[n];
    #pragma unroll
    for (int mt = 0; mt < 4; ++mt) {
      #pragma unroll
      for (int r = 0; r < 8; ++r) {
        const int m = bm + wm * 64 + mt * 16 + r + 8 * lkg;
        float val = fmaxf(acc[mt][nt][r] + bv, 0.0f);
        if constexpr (OUT_MODE == 0) {
          Oh[(size_t)m * N + n] = bf16_rtne(val);
        } else if constexpr (OUT_MODE == 1) {
          const int bb = m >> 11, ss = m & (Ss - 1);
          const int hh = n >> 6,  dd = n & 63;
          Oh[(((size_t)bb * Hh + hh) * Dd + dd) * Ss + ss] = bf16_rtne(val);
        } else {
          Of[(size_t)m * N + n] = val;
        }
      }
    }
  }
}

// One workgroup (4 waves / 128 threads) per (b, h, 16-query tile).
// LDS: sc[16][2048] f32 (128KB) + at[16][2048] bf16 (64KB) + reductions.
__global__ __launch_bounds__(128)
void attn_fused(const unsigned short* __restrict__ qh,
                const unsigned short* __restrict__ kh,
                const unsigned short* __restrict__ vt,   // [B][H][D][S]
                const float* __restrict__ mask,
                const float* __restrict__ mask2,
                unsigned short* __restrict__ oh) {
  extern __shared__ char smem[];
  float* sc          = (float*)smem;                              // 16*2048 f32
  unsigned short* at = (unsigned short*)(smem + 16 * Ss * 4);     // 16*2048 bf16
  float* red         = (float*)(smem + 16 * Ss * 4 + 16 * Ss * 2);
  float* rowm        = red + 128;
  float* rowr        = rowm + 16;

  const int tid  = threadIdx.x;
  const int lane = tid & 31;
  const int wave = tid >> 5;
  const int lrow = lane & 15;
  const int lkg  = lane >> 4;

  // h fastest so the 16 heads sharing mask rows hit L2 together
  const int h  = blockIdx.x & (Hh - 1);
  const int qt = (blockIdx.x >> 4) & 127;
  const int b  = blockIdx.x >> 11;
  const int qbase    = qt * 16;
  const size_t mrow0 = (size_t)b * Ss + qbase;

  // Q A-fragments held for the whole kernel (K=64 -> two k-steps of 32)
  v16u aq[2];
  {
    const unsigned short* qp = qh + ((size_t)b * Ss + qbase + lrow) * Ee + h * Dd;
    #pragma unroll
    for (int ks = 0; ks < 2; ++ks) {
      const unsigned short* p = qp + ks * 32 + lkg * 8;
      aq[ks] = cat16(*(const v8u*)p, *(const v8u*)(p + 16));
    }
  }

  const float scale = 0.125f;  // 1/sqrt(64)

  // Pass 1: scores = scale * q @ k^T for keys [wave*512, wave*512+512)
  for (int kt = 0; kt < 32; ++kt) {
    const int t0 = wave * 512 + kt * 16;
    v8f acc = {};
    const unsigned short* kp = kh + ((size_t)b * Ss + t0 + lrow) * Ee + h * Dd;
    #pragma unroll
    for (int ks = 0; ks < 2; ++ks) {
      const unsigned short* p = kp + ks * 32 + lkg * 8;
      v16u bfr = cat16(*(const v8u*)p, *(const v8u*)(p + 16));
      acc = wmma_bf16(aq[ks], bfr, acc);
    }
    #pragma unroll
    for (int r = 0; r < 8; ++r)
      sc[(r + 8 * lkg) * Ss + t0 + lrow] = acc[r] * scale;
  }
  __syncthreads();

  // Pass 2: add pre-softmax mask, partial row-max
  {
    const int row = tid >> 3, s8 = tid & 7;
    const float* mrow = mask + (mrow0 + row) * Ss;
    float lmax = -3.4e38f;
    for (int j = 0; j < 256; ++j) {
      const int t = j * 8 + s8;
      float s = sc[row * Ss + t] + mrow[t];
      sc[row * Ss + t] = s;
      lmax = fmaxf(lmax, s);
    }
    red[row * 8 + s8] = lmax;
  }
  __syncthreads();
  if (tid < 16) {
    float m = red[tid * 8];
    for (int j = 1; j < 8; ++j) m = fmaxf(m, red[tid * 8 + j]);
    rowm[tid] = m;
  }
  __syncthreads();

  // Pass 3: row sum of exp
  {
    const int row = tid >> 3, s8 = tid & 7;
    const float m = rowm[row];
    float lsum = 0.0f;
    for (int j = 0; j < 256; ++j)
      lsum += __expf(sc[row * Ss + j * 8 + s8] - m);
    red[row * 8 + s8] = lsum;
  }
  __syncthreads();
  if (tid < 16) {
    float s = 0.0f;
    for (int j = 0; j < 8; ++j) s += red[tid * 8 + j];
    rowr[tid] = 1.0f / s;
  }
  __syncthreads();

  // Pass 4: attn = relu(softmax + mask2) -> bf16 in LDS (coalesced sweep)
  {
    const float* m2b = mask2 + mrow0 * Ss;
    for (int e = tid; e < 16 * Ss; e += 128) {
      const int row = e >> 11;
      const int t   = e & (Ss - 1);
      float p = __expf(sc[e] - rowm[row]) * rowr[row] + m2b[(size_t)row * Ss + t];
      at[e] = bf16_rtne(fmaxf(p, 0.0f));
    }
  }
  __syncthreads();

  // Pass 5: out = attn @ V ; wave owns d-slice [wave*16, wave*16+16)
  v8f acc = {};
  const unsigned short* vrow =
      vt + (((size_t)b * Hh + h) * Dd + wave * 16 + lrow) * Ss;
  for (int ks = 0; ks < 64; ++ks) {
    const unsigned short* ap = at + lrow * Ss + ks * 32 + lkg * 8;
    v16u af = cat16(*(const v8u*)ap, *(const v8u*)(ap + 16));
    const unsigned short* bp = vrow + ks * 32 + lkg * 8;
    v16u bf = cat16(*(const v8u*)bp, *(const v8u*)(bp + 16));
    acc = wmma_bf16(af, bf, acc);
  }
  #pragma unroll
  for (int r = 0; r < 8; ++r) {
    const int m = qbase + r + 8 * lkg;
    oh[((size_t)b * Ss + m) * Ee + h * Dd + wave * 16 + lrow] = bf16_rtne(acc[r]);
  }
}

extern "C" void kernel_launch(void* const* d_in, const int* in_sizes, int n_in,
                              void* d_out, int out_size, void* d_ws, size_t ws_size,
                              hipStream_t stream) {
  (void)in_sizes; (void)n_in; (void)out_size; (void)ws_size;
  const float* q     = (const float*)d_in[0];
  const float* k     = (const float*)d_in[1];
  const float* v     = (const float*)d_in[2];
  const float* mask  = (const float*)d_in[3];
  const float* mask2 = (const float*)d_in[4];
  const float* Wq    = (const float*)d_in[5];
  const float* bq    = (const float*)d_in[6];
  const float* Wk    = (const float*)d_in[7];
  const float* bk    = (const float*)d_in[8];
  const float* Wv    = (const float*)d_in[9];
  const float* bv    = (const float*)d_in[10];
  const float* Wo    = (const float*)d_in[11];
  const float* bo    = (const float*)d_in[12];
  float* out = (float*)d_out;

  const size_t tok  = (size_t)Bb * Ss * Ee;  // 4M elements
  const size_t wsz  = (size_t)Ee * Ee;       // 1M elements
  unsigned short* qh  = (unsigned short*)d_ws;
  unsigned short* kh  = qh + tok;
  unsigned short* vt  = kh + tok;
  unsigned short* ohb = vt + tok;
  unsigned short* xb  = ohb + tok;  // bf16 input staging (reused q/k/v)
  unsigned short* wb  = xb + tok;   // bf16 weight staging (reused)

  const int gx = (int)(tok / 4 / 256);  // 4096
  const int gw = (int)(wsz / 4 / 256);  // 1024
  dim3 gp(Bb * Ss / 128, Ee / 128);     // 32 x 8

  cvt_f32_bf16<<<gx, 256, 0, stream>>>(q, xb, (int)(tok / 4));
  cvt_f32_bf16<<<gw, 256, 0, stream>>>(Wq, wb, (int)(wsz / 4));
  proj_gemm<0><<<gp, 256, 0, stream>>>(xb, wb, bq, qh);

  cvt_f32_bf16<<<gx, 256, 0, stream>>>(k, xb, (int)(tok / 4));
  cvt_f32_bf16<<<gw, 256, 0, stream>>>(Wk, wb, (int)(wsz / 4));
  proj_gemm<0><<<gp, 256, 0, stream>>>(xb, wb, bk, kh);

  cvt_f32_bf16<<<gx, 256, 0, stream>>>(v, xb, (int)(tok / 4));
  cvt_f32_bf16<<<gw, 256, 0, stream>>>(Wv, wb, (int)(wsz / 4));
  proj_gemm<1><<<gp, 256, 0, stream>>>(xb, wb, bv, vt);

  const size_t attn_lds = (size_t)16 * Ss * 4 + (size_t)16 * Ss * 2 + 160 * 4;
  attn_fused<<<Bb * Hh * (Ss / 16), 128, attn_lds, stream>>>(qh, kh, vt, mask,
                                                             mask2, ohb);

  cvt_f32_bf16<<<gw, 256, 0, stream>>>(Wo, wb, (int)(wsz / 4));
  proj_gemm<2><<<gp, 256, 0, stream>>>(ohb, wb, bo, out);
}
